// MultiHeadSelfAttention_3891240370698
// MI455X (gfx1250) — matrix operand order repaired and checked by reading, not compiled, not run on hardware
//
#include <hip/hip_runtime.h>

// MI455X (gfx1250) multi-head self-attention, bf16 WMMA path, fp32 accumulate.
// B=4, L=2048, D=1024, H=16, hd=64.

#define BB 4
#define LL 2048
#define DD 1024
#define HH 16
#define HD 64
#define MTOT (BB * LL) // 8192

typedef __attribute__((ext_vector_type(8)))  __bf16 v8bf;
typedef __attribute__((ext_vector_type(16))) __bf16 v16bf;
typedef __attribute__((ext_vector_type(8)))  float  v8f;

__device__ __forceinline__ v16bf cat8(v8bf lo, v8bf hi) {
    return __builtin_shufflevector(lo, hi, 0, 1, 2, 3, 4, 5, 6, 7,
                                          8, 9, 10, 11, 12, 13, 14, 15);
}

__device__ __forceinline__ v8f wmma_bf16(v16bf a, v16bf b, v8f c) {
    return __builtin_amdgcn_wmma_f32_16x16x32_bf16(false, a, false, b,
                                                   (short)0, c, false, false);
}

// ---------------------------------------------------------------- convert ---
__global__ __launch_bounds__(256) void cvt_f32_bf16(const float* __restrict__ src,
                                                    __bf16* __restrict__ dst, int n) {
    int i = (blockIdx.x * 256 + threadIdx.x) * 4;
    if (i + 3 < n) {
        float4 v = *(const float4*)(src + i);
        dst[i + 0] = (__bf16)v.x;
        dst[i + 1] = (__bf16)v.y;
        dst[i + 2] = (__bf16)v.z;
        dst[i + 3] = (__bf16)v.w;
    }
}

// --------------------------------------------- QKV projection + RoPE (bf16) ---
// grid = (D/64=16, MTOT/128=64, 3), block = 256 (8 waves).
// Each wave: 16 rows x 64 cols, K-loop over 1024 in steps of 32.
// Double-buffered LDS W-tile: one barrier per K step; staging of buf^1
// overlaps WMMAs on buf.
// z=0 -> Q (RoPE, [b,h,l,d]); z=1 -> K (RoPE, [b,h,l,d]); z=2 -> V ([b,h,d,l]).
__global__ __launch_bounds__(256)
void qkv_rope_kernel(const __bf16* __restrict__ X,
                     const __bf16* __restrict__ Wq, const __bf16* __restrict__ Wk,
                     const __bf16* __restrict__ Wv,
                     __bf16* __restrict__ Qo, __bf16* __restrict__ Ko,
                     __bf16* __restrict__ Vo) {
    const int tid   = threadIdx.x;
    const int wave  = tid >> 5;
    const int lane  = tid & 31;
    const int lhalf = lane & 15;
    const bool hi   = lane >= 16;
    const int z     = blockIdx.z;
    const __bf16* W = (z == 0) ? Wq : ((z == 1) ? Wk : Wv);

    const int m_base = blockIdx.y * 128 + wave * 16;
    const int n_base = blockIdx.x * 64;

    __shared__ __align__(16) __bf16 ldsB[2][64 * 32]; // 2 x 4 KB W-tiles

    const int sr = tid >> 2, sc = (tid & 3) * 8; // cooperative-stage coords
    const __bf16* wsrc = W + (n_base + sr) * DD + sc;

    v8f acc0 = {}, acc1 = {}, acc2 = {}, acc3 = {};

    // preload first tile
    *(v8bf*)(ldsB[0] + sr * 32 + sc) = *(const v8bf*)(wsrc);
    __syncthreads();

    for (int k0 = 0; k0 < DD; k0 += 32) {
        const int cur = (k0 >> 5) & 1;
        if (k0 + 32 < DD) // stage next tile into the other buffer (no barrier needed)
            *(v8bf*)(ldsB[cur ^ 1] + sr * 32 + sc) = *(const v8bf*)(wsrc + k0 + 32);

        // A fragment 16x32 (bf16): lanes<16 hold K{0..15}, lanes>=16 K{16..31}
        const __bf16* xrow = X + (m_base + lhalf) * DD + k0 + (hi ? 16 : 0);
        v16bf afrag = cat8(*(const v8bf*)(xrow), *(const v8bf*)(xrow + 8));
        if (k0 + 32 < DD) __builtin_prefetch(xrow + 32, 0, 3);

        const __bf16* bb = ldsB[cur] + (hi ? 16 : 0);
        v16bf b0 = *(const v16bf*)(bb + (0 * 16 + lhalf) * 32);
        v16bf b1 = *(const v16bf*)(bb + (1 * 16 + lhalf) * 32);
        v16bf b2 = *(const v16bf*)(bb + (2 * 16 + lhalf) * 32);
        v16bf b3 = *(const v16bf*)(bb + (3 * 16 + lhalf) * 32);

        acc0 = wmma_bf16(afrag, b0, acc0);
        acc1 = wmma_bf16(afrag, b1, acc1);
        acc2 = wmma_bf16(afrag, b2, acc2);
        acc3 = wmma_bf16(afrag, b3, acc3);
        __syncthreads(); // next iter stages over buffer `cur^1`... reads of `cur` done
    }

    const int b = m_base >> 11; // 128-row block lies inside one batch
    const int h = blockIdx.x;   // one 64-wide N tile == one head

    if (z < 2) {
        // RoPE: col j = t*16+lhalf; rotate_half partner is acc[t^2], same component.
        float inv0 = __powf(10000.0f, -((float)lhalf) / 32.0f);
        float inv1 = __powf(10000.0f, -((float)(lhalf + 16)) / 32.0f);
        __bf16* Out = (z == 0) ? Qo : Ko;
#pragma unroll
        for (int c = 0; c < 8; ++c) {
            int row = m_base + c + (hi ? 8 : 0);
            int pos = row & (LL - 1);
            float s0, c0, s1, c1;
            __sincosf((float)pos * inv0, &s0, &c0);
            __sincosf((float)pos * inv1, &s1, &c1);
            float q0 = acc0[c], q1 = acc1[c], q2 = acc2[c], q3 = acc3[c];
            float r0 = q0 * c0 - q2 * s0;
            float r1 = q1 * c1 - q3 * s1;
            float r2 = q2 * c0 + q0 * s0;
            float r3 = q3 * c1 + q1 * s1;
            __bf16* dst = Out + ((size_t)((b * HH + h) * LL + pos)) * HD;
            dst[0 * 16 + lhalf] = (__bf16)r0;
            dst[1 * 16 + lhalf] = (__bf16)r1;
            dst[2 * 16 + lhalf] = (__bf16)r2;
            dst[3 * 16 + lhalf] = (__bf16)r3;
        }
    } else {
        // V stored transposed: [b,h,d,l] so PV B-fragments are contiguous in l.
#pragma unroll
        for (int c = 0; c < 8; ++c) {
            int row = m_base + c + (hi ? 8 : 0);
            int pos = row & (LL - 1);
            __bf16* dst = Vo + ((size_t)((b * HH + h) * HD)) * LL + pos;
            dst[(0 * 16 + lhalf) * LL] = (__bf16)acc0[c];
            dst[(1 * 16 + lhalf) * LL] = (__bf16)acc1[c];
            dst[(2 * 16 + lhalf) * LL] = (__bf16)acc2[c];
            dst[(3 * 16 + lhalf) * LL] = (__bf16)acc3[c];
        }
    }
}

// ----------------------------------------------------- flash attention (bf16) ---
// grid = (L/128=16, B*H=64), block = 256. Each wave: 16 query rows, online
// softmax over 64-key blocks. Waves fully independent (no block barriers).
__global__ __launch_bounds__(256)
void attn_kernel(const __bf16* __restrict__ Q, const __bf16* __restrict__ K,
                 const __bf16* __restrict__ Vt, __bf16* __restrict__ O) {
    const int tid   = threadIdx.x;
    const int wave  = tid >> 5;
    const int lane  = tid & 31;
    const int lhalf = lane & 15;
    const bool hi   = lane >= 16;
    const int bh    = blockIdx.y;
    const int qbase = blockIdx.x * 128 + wave * 16;

    const __bf16* Qh = Q + (size_t)bh * LL * HD;
    const __bf16* Kh = K + (size_t)bh * LL * HD;
    const __bf16* Vh = Vt + (size_t)bh * HD * LL;

    __shared__ __align__(16) __bf16 ldsP[8][16 * 64]; // 2 KB wave-private P tile
    __bf16* P = ldsP[wave];

    // Q A-fragments (d=0..31 and d=32..63), loaded once.
    const __bf16* qrow = Qh + (qbase + lhalf) * HD + (hi ? 16 : 0);
    v16bf aq0 = cat8(*(const v8bf*)(qrow), *(const v8bf*)(qrow + 8));
    v16bf aq1 = cat8(*(const v8bf*)(qrow + 32), *(const v8bf*)(qrow + 40));

    v8f o0 = {}, o1 = {}, o2 = {}, o3 = {};
    float rmax[8], rsum[8];
#pragma unroll
    for (int c = 0; c < 8; ++c) { rmax[c] = -1e30f; rsum[c] = 0.0f; }

    for (int j0 = 0; j0 < LL; j0 += 64) {
        // S = (Q K^T) * 1/sqrt(64): four 16x16 tiles
        v8f s[4];
#pragma unroll
        for (int t = 0; t < 4; ++t) {
            const __bf16* krow = Kh + (j0 + t * 16 + lhalf) * HD;
            v16bf b0 = *(const v16bf*)(krow + (hi ? 16 : 0));
            v16bf b1 = *(const v16bf*)(krow + 32 + (hi ? 16 : 0));
            v8f zz = {};
            zz = wmma_bf16(aq0, b0, zz);
            zz = wmma_bf16(aq1, b1, zz);
            s[t] = zz * 0.125f;
        }
        if (j0 + 64 < LL) __builtin_prefetch(Kh + (j0 + 64 + lhalf) * HD, 0, 3);

        // online softmax per row (row = c + 8*hi; cols live in 16-lane halves)
        float scl[8];
#pragma unroll
        for (int c = 0; c < 8; ++c) {
            float m = fmaxf(fmaxf(s[0][c], s[1][c]), fmaxf(s[2][c], s[3][c]));
#pragma unroll
            for (int off = 1; off < 16; off <<= 1) m = fmaxf(m, __shfl_xor(m, off, 32));
            float mnew = fmaxf(rmax[c], m);
            float corr = __expf(rmax[c] - mnew);
            rmax[c] = mnew;
            float psum = 0.0f;
            const int prow = (c + (hi ? 8 : 0)) * 64 + lhalf;
#pragma unroll
            for (int t = 0; t < 4; ++t) {
                float p = __expf(s[t][c] - mnew);
                psum += p;
                P[prow + t * 16] = (__bf16)p;
            }
#pragma unroll
            for (int off = 1; off < 16; off <<= 1) psum += __shfl_xor(psum, off, 32);
            rsum[c] = rsum[c] * corr + psum;
            scl[c] = corr;
        }
#pragma unroll
        for (int c = 0; c < 8; ++c) {
            o0[c] *= scl[c]; o1[c] *= scl[c]; o2[c] *= scl[c]; o3[c] *= scl[c];
        }

        // same-wave LDS is in-order; drain DS stores before the cross-lane re-read
        asm volatile("s_wait_dscnt 0" ::: "memory");

        // P A-fragments from LDS (keys 0..31 and 32..63 of this block)
        const __bf16* pr = P + lhalf * 64 + (hi ? 16 : 0);
        v16bf ap0 = cat8(*(const v8bf*)(pr), *(const v8bf*)(pr + 8));
        v16bf ap1 = cat8(*(const v8bf*)(pr + 32), *(const v8bf*)(pr + 40));

        // O += P V : V^T rows are contiguous over keys
        {
            const __bf16* vr0 = Vh + (0 * 16 + lhalf) * LL + j0 + (hi ? 16 : 0);
            const __bf16* vr1 = Vh + (1 * 16 + lhalf) * LL + j0 + (hi ? 16 : 0);
            const __bf16* vr2 = Vh + (2 * 16 + lhalf) * LL + j0 + (hi ? 16 : 0);
            const __bf16* vr3 = Vh + (3 * 16 + lhalf) * LL + j0 + (hi ? 16 : 0);
            o0 = wmma_bf16(ap0, *(const v16bf*)(vr0), o0);
            o1 = wmma_bf16(ap0, *(const v16bf*)(vr1), o1);
            o2 = wmma_bf16(ap0, *(const v16bf*)(vr2), o2);
            o3 = wmma_bf16(ap0, *(const v16bf*)(vr3), o3);
            o0 = wmma_bf16(ap1, *(const v16bf*)(vr0 + 32), o0);
            o1 = wmma_bf16(ap1, *(const v16bf*)(vr1 + 32), o1);
            o2 = wmma_bf16(ap1, *(const v16bf*)(vr2 + 32), o2);
            o3 = wmma_bf16(ap1, *(const v16bf*)(vr3 + 32), o3);
        }
        // compiler fence only: same-wave DS ordering guarantees the next
        // iteration's P stores cannot bypass the ap0/ap1 loads.
        asm volatile("" ::: "memory");
    }

    // finalize: O /= rowsum, store bf16 to [B, L, D] (heads re-interleaved)
    const int b = bh >> 4, h = bh & 15;
#pragma unroll
    for (int c = 0; c < 8; ++c) {
        int qq = qbase + c + (hi ? 8 : 0);
        float inv = 1.0f / rsum[c];
        __bf16* dst = O + ((size_t)(b * LL + qq)) * DD + h * HD;
        dst[0 * 16 + lhalf] = (__bf16)(o0[c] * inv);
        dst[1 * 16 + lhalf] = (__bf16)(o1[c] * inv);
        dst[2 * 16 + lhalf] = (__bf16)(o2[c] * inv);
        dst[3 * 16 + lhalf] = (__bf16)(o3[c] * inv);
    }
}

// -------------------------------------------------- output projection (fp32) ---
// grid = (16, 64), block 256. out = A @ Wo^T, fp32 stores to d_out.
// Same double-buffered tiling as the QKV GEMM.
__global__ __launch_bounds__(256)
void oproj_kernel(const __bf16* __restrict__ A, const __bf16* __restrict__ W,
                  float* __restrict__ out) {
    const int tid   = threadIdx.x;
    const int wave  = tid >> 5;
    const int lane  = tid & 31;
    const int lhalf = lane & 15;
    const bool hi   = lane >= 16;
    const int m_base = blockIdx.y * 128 + wave * 16;
    const int n_base = blockIdx.x * 64;

    __shared__ __align__(16) __bf16 ldsB[2][64 * 32];

    const int sr = tid >> 2, sc = (tid & 3) * 8;
    const __bf16* wsrc = W + (n_base + sr) * DD + sc;

    v8f acc0 = {}, acc1 = {}, acc2 = {}, acc3 = {};

    *(v8bf*)(ldsB[0] + sr * 32 + sc) = *(const v8bf*)(wsrc);
    __syncthreads();

    for (int k0 = 0; k0 < DD; k0 += 32) {
        const int cur = (k0 >> 5) & 1;
        if (k0 + 32 < DD)
            *(v8bf*)(ldsB[cur ^ 1] + sr * 32 + sc) = *(const v8bf*)(wsrc + k0 + 32);

        const __bf16* arow = A + (m_base + lhalf) * DD + k0 + (hi ? 16 : 0);
        v16bf afrag = cat8(*(const v8bf*)(arow), *(const v8bf*)(arow + 8));
        if (k0 + 32 < DD) __builtin_prefetch(arow + 32, 0, 3);

        const __bf16* bb = ldsB[cur] + (hi ? 16 : 0);
        v16bf b0 = *(const v16bf*)(bb + (0 * 16 + lhalf) * 32);
        v16bf b1 = *(const v16bf*)(bb + (1 * 16 + lhalf) * 32);
        v16bf b2 = *(const v16bf*)(bb + (2 * 16 + lhalf) * 32);
        v16bf b3 = *(const v16bf*)(bb + (3 * 16 + lhalf) * 32);

        acc0 = wmma_bf16(afrag, b0, acc0);
        acc1 = wmma_bf16(afrag, b1, acc1);
        acc2 = wmma_bf16(afrag, b2, acc2);
        acc3 = wmma_bf16(afrag, b3, acc3);
        __syncthreads();
    }

#pragma unroll
    for (int c = 0; c < 8; ++c) {
        size_t row = (size_t)(m_base + c + (hi ? 8 : 0));
        float* dst = out + row * DD + n_base;
        dst[0 * 16 + lhalf] = acc0[c];
        dst[1 * 16 + lhalf] = acc1[c];
        dst[2 * 16 + lhalf] = acc2[c];
        dst[3 * 16 + lhalf] = acc3[c];
    }
}

// ------------------------------------------------------------------- launch ---
extern "C" void kernel_launch(void* const* d_in, const int* in_sizes, int n_in,
                              void* d_out, int out_size, void* d_ws, size_t ws_size,
                              hipStream_t stream) {
    const float* x  = (const float*)d_in[0];
    const float* Wq = (const float*)d_in[1];
    const float* Wk = (const float*)d_in[2];
    const float* Wv = (const float*)d_in[3];
    const float* Wo = (const float*)d_in[4];

    __bf16* ws = (__bf16*)d_ws;
    size_t off = 0;
    __bf16* Xb  = ws + off; off += (size_t)MTOT * DD;      // 8.4M
    __bf16* Wqb = ws + off; off += (size_t)DD * DD;        // 1M
    __bf16* Wkb = ws + off; off += (size_t)DD * DD;
    __bf16* Wvb = ws + off; off += (size_t)DD * DD;
    __bf16* Wob = ws + off; off += (size_t)DD * DD;
    __bf16* Qb  = ws + off; off += (size_t)BB * HH * LL * HD; // [b,h,l,d]
    __bf16* Kb  = ws + off; off += (size_t)BB * HH * LL * HD; // [b,h,l,d]
    __bf16* Vtb = ws + off; off += (size_t)BB * HH * HD * LL; // [b,h,d,l]
    __bf16* Ob  = ws + off; off += (size_t)MTOT * DD;         // [b,l,D]

    const int nx = MTOT * DD;
    const int nw = DD * DD;
    cvt_f32_bf16<<<nx / (256 * 4), 256, 0, stream>>>(x, Xb, nx);
    cvt_f32_bf16<<<nw / (256 * 4), 256, 0, stream>>>(Wq, Wqb, nw);
    cvt_f32_bf16<<<nw / (256 * 4), 256, 0, stream>>>(Wk, Wkb, nw);
    cvt_f32_bf16<<<nw / (256 * 4), 256, 0, stream>>>(Wv, Wvb, nw);
    cvt_f32_bf16<<<nw / (256 * 4), 256, 0, stream>>>(Wo, Wob, nw);

    dim3 g2(DD / 64, MTOT / 128, 3);
    qkv_rope_kernel<<<g2, 256, 0, stream>>>(Xb, Wqb, Wkb, Wvb, Qb, Kb, Vtb);

    dim3 g3(LL / 128, BB * HH);
    attn_kernel<<<g3, 256, 0, stream>>>(Qb, Kb, Vtb, Ob);

    dim3 g4(DD / 64, MTOT / 128);
    oproj_kernel<<<g4, 256, 0, stream>>>(Ob, Wob, (float*)d_out);
}
